// LSTMClassifier_15015205667166
// MI455X (gfx1250) — compile-verified, matrix operand order checked by reading, not run on hardware
//
#include <hip/hip_runtime.h>
#include <cstdint>
#include <cstddef>

// Problem sizes (fixed by the reference)
#define B_SZ 64
#define T_SZ 512
#define I_SZ 256
#define H_SZ 512
#define G_SZ 2048            // 4*H (gate columns, PyTorch order i,f,g,o)
#define C_SZ 1000
#define BT   (B_SZ * T_SZ)   // 32768
#define SCAN_WGS 32          // persistent workgroups in the recurrent kernel

typedef __bf16 bf16_t;
typedef __attribute__((ext_vector_type(16))) __bf16 v16bf;
typedef __attribute__((ext_vector_type(8)))  __bf16 v8bf;
typedef __attribute__((ext_vector_type(8)))  float  v8f;

// ---------------------------------------------------------------------------
// WMMA helpers (wave32, v_wmma_f32_16x16x32_bf16)
//
// A (16x32 bf16) per-lane layout (ISA 7.12.2): lanes 0-15 hold M=lane,
// lanes 16-31 hold M=lane-16. VGPR0..3: K = kb..kb+7, VGPR4..7: K = 16+kb..+7
// with kb = 0 (lanes 0-15) or 8 (lanes 16-31) -> two contiguous 16B chunks.
// B (32x16) is loaded with the same per-lane pattern from the weight rows
// (B[k][n] = W[n][k], column-major B == W row-major, contiguous in K).
// ---------------------------------------------------------------------------
static __device__ __forceinline__ v16bf load_frag(const bf16_t* __restrict__ tile,
                                                  int ld, int k0, int lane) {
  const int r  = lane & 15;
  const int kb = (lane & 16) ? 8 : 0;
  const bf16_t* q = tile + (size_t)r * (size_t)ld + (size_t)(k0 + kb);
  union { v16bf v; v8bf h[2]; } u;
  u.h[0] = *(const v8bf*)(q);        // K = k0+kb .. +7
  u.h[1] = *(const v8bf*)(q + 16);   // K = k0+16+kb .. +7
  return u.v;
}

static __device__ __forceinline__ v8f wmma_bf16(v16bf a, v16bf b, v8f c) {
  return __builtin_amdgcn_wmma_f32_16x16x32_bf16(false, a, false, b,
                                                 (short)0, c, false, false);
}

static __device__ __forceinline__ float sigmoidf_(float x) {
  return 1.0f / (1.0f + __expf(-x));
}

// ---------------------------------------------------------------------------
// fp32 -> bf16 conversion + bias combine
// ---------------------------------------------------------------------------
__global__ void cvt_f32_bf16(const float* __restrict__ in, bf16_t* __restrict__ out, int n) {
  int i = blockIdx.x * blockDim.x + threadIdx.x;
  if (i < n) out[i] = (bf16_t)in[i];
}

__global__ void bias_sum(const float* __restrict__ a, const float* __restrict__ b,
                         float* __restrict__ o, int n) {
  int i = blockIdx.x * blockDim.x + threadIdx.x;
  if (i < n) o[i] = a[i] + b[i];
}

// ---------------------------------------------------------------------------
// Input GEMM: xg = A[BT, K] @ W[4H, K]^T + bias, stored TRANSPOSED as
// xg[t][b][4H] so the recurrent scan streams one contiguous 512KB block per
// timestep. One wave computes a 32x64 tile (8 accumulators). K is a compile-
// time constant (256 or 512): the k-loop is fully unrolled into one DAG so
// the scheduler hoists global loads across WMMAs with no loop-edge copies.
// ---------------------------------------------------------------------------
template <int K>
__global__ void __launch_bounds__(128)
gemm_xg(const bf16_t* __restrict__ A, const bf16_t* __restrict__ W,
        const float* __restrict__ bias, float* __restrict__ Y) {
  const int lane  = threadIdx.x & 31;
  const int wave  = blockIdx.x * 4 + (threadIdx.x >> 5);
  const int tilesN = G_SZ / 64;                 // 32
  const int m0 = (wave / tilesN) << 5;          // 32-row tile, < BT
  const int n0 = (wave % tilesN) << 6;

  const bf16_t* At0 = A + (size_t)m0 * (size_t)K;
  const bf16_t* At1 = A + (size_t)(m0 + 16) * (size_t)K;
  const bf16_t* Wt[4];
#pragma unroll
  for (int nt = 0; nt < 4; ++nt)
    Wt[nt] = W + (size_t)(n0 + nt * 16) * (size_t)K;

  v8f acc[2][4] = {};
#pragma unroll
  for (int k0 = 0; k0 < K; k0 += 32) {
    const v16bf a0 = load_frag(At0, K, k0, lane);
    const v16bf a1 = load_frag(At1, K, k0, lane);
#pragma unroll
    for (int nt = 0; nt < 4; ++nt) {
      const v16bf b = load_frag(Wt[nt], K, k0, lane);
      acc[0][nt] = wmma_bf16(a0, b, acc[0][nt]);
      acc[1][nt] = wmma_bf16(a1, b, acc[1][nt]);
    }
  }

  // C/D layout: VGPR r -> M = r + 8*(lane>=16), N = lane&15.
  // Store transposed: A-row m = bb*T + tt  ->  Y[(tt*B + bb)*4H + n].
  const int nn = lane & 15;
#pragma unroll
  for (int mh = 0; mh < 2; ++mh) {
    const int mr = m0 + mh * 16 + ((lane >> 4) << 3);
#pragma unroll
    for (int nt = 0; nt < 4; ++nt) {
      const int n = n0 + nt * 16 + nn;
      const float bv = bias[n];
#pragma unroll
      for (int r = 0; r < 8; ++r) {
        const int m  = mr + r;
        const int bb = m >> 9;           // T = 512
        const int tt = m & (T_SZ - 1);
        Y[((size_t)tt * B_SZ + bb) * G_SZ + n] = acc[mh][nt][r] + bv;
      }
    }
  }
}

// ---------------------------------------------------------------------------
// Device-wide generation barrier for the persistent scan kernel.
// bar[0] = arrive counter, bar[1] = generation. Reset by hipMemsetAsync.
// ---------------------------------------------------------------------------
static __device__ __forceinline__ void grid_sync_(unsigned* cnt, unsigned* gen) {
  __threadfence();
  __syncthreads();
  if (threadIdx.x == 0) {
    unsigned old = atomicAdd(gen, 0u);          // read generation BEFORE arriving
    if (atomicAdd(cnt, 1u) == SCAN_WGS - 1u) {
      atomicExch(cnt, 0u);
      __threadfence();
      atomicAdd(gen, 1u);
    } else {
      while (atomicAdd(gen, 0u) == old) __builtin_amdgcn_s_sleep(2);
    }
    __threadfence();
  }
  __syncthreads();
}

// ---------------------------------------------------------------------------
// Persistent recurrent scan for one LSTM layer.
// 32 WGs x 128 threads (4 waves). WG owns hidden units [j0, j0+16); wave g
// computes gate g's 64x16 pre-activation tile via WMMA over K=H. The k-loop
// (16 chunks, compile-time) is fully unrolled: one straight-line DAG per
// timestep, loads free to overlap WMMAs. Cell state lives in LDS; h is
// double-buffered in global (bf16) with a device-wide barrier per timestep.
// ---------------------------------------------------------------------------
__global__ void __launch_bounds__(128)
lstm_scan(const bf16_t* __restrict__ Whh,   // [4H, H] bf16
          const float*  __restrict__ xg,    // [T, B, 4H] fp32 (bias folded in)
          bf16_t* __restrict__ hseq,        // [BT, H] bf16 layer output (b-major)
          bf16_t* __restrict__ hbuf,        // [2, B, H] bf16 double buffer
          unsigned* __restrict__ bar) {
  __shared__ float glds[4][B_SZ][16];   // gate exchange (16 KB)
  __shared__ float clds[B_SZ][16];      // persistent cell state (4 KB)

  const int tid  = threadIdx.x;
  const int lane = tid & 31;
  const int g    = tid >> 5;            // gate index = wave id (i,f,g,o)
  const int j0   = blockIdx.x << 4;     // hidden-unit slice

  // init: c = 0, h(buffer 0) = 0 for this WG's slice
  for (int idx = tid; idx < B_SZ * 16; idx += 128) {
    const int b = idx >> 4, j = idx & 15;
    clds[b][j] = 0.0f;
    hbuf[(size_t)b * H_SZ + j0 + j] = (bf16_t)0.0f;
  }
  grid_sync_(bar, bar + 1);

  const bf16_t* Wrow = Whh + (size_t)(g * H_SZ + j0) * H_SZ;

  for (int t = 0; t < T_SZ; ++t) {
    const bf16_t* hin = hbuf + (size_t)(t & 1) * (B_SZ * H_SZ);
    v8f acc[4] = {};
#pragma unroll
    for (int k0 = 0; k0 < H_SZ; k0 += 32) {
      const v16bf b = load_frag(Wrow, H_SZ, k0, lane);   // L2-resident weights
#pragma unroll
      for (int mt = 0; mt < 4; ++mt) {
        const v16bf a = load_frag(hin + (size_t)(mt * 16) * H_SZ, H_SZ, k0, lane);
        acc[mt] = wmma_bf16(a, b, acc[mt]);
      }
    }

    // add precomputed input contribution (contiguous [B,4H] block at step t),
    // exchange gates through LDS
    const float* xgt = xg + (size_t)t * B_SZ * G_SZ;
    const int nn = lane & 15;
    const int gc = g * H_SZ + j0 + nn;
    const int mb = (lane >> 4) << 3;
#pragma unroll
    for (int mt = 0; mt < 4; ++mt) {
#pragma unroll
      for (int r = 0; r < 8; ++r) {
        const int b = mt * 16 + mb + r;
        glds[g][b][nn] = acc[mt][r] + xgt[(size_t)b * G_SZ + gc];
      }
    }
    __syncthreads();

    // pointwise gate math + state update (fp32)
    bf16_t* hout = hbuf + (size_t)((t + 1) & 1) * (B_SZ * H_SZ);
    for (int idx = tid; idx < B_SZ * 16; idx += 128) {
      const int b = idx >> 4, j = idx & 15;
      const float iv = sigmoidf_(glds[0][b][j]);
      const float fv = sigmoidf_(glds[1][b][j]);
      const float gv = tanhf(glds[2][b][j]);
      const float ov = sigmoidf_(glds[3][b][j]);
      const float c  = fv * clds[b][j] + iv * gv;
      clds[b][j] = c;
      const float h = ov * tanhf(c);
      const bf16_t hb = (bf16_t)h;
      hout[(size_t)b * H_SZ + j0 + j] = hb;
      hseq[((size_t)b * T_SZ + t) * H_SZ + j0 + j] = hb;
    }
    grid_sync_(bar, bar + 1);   // publish h for all WGs before step t+1
  }
}

// ---------------------------------------------------------------------------
// Head: out[64,1000] = h_last[64,512] @ W_fc[1000,512]^T + b_fc
// W_fc rows zero-padded to 1008 so fragment loads need no guards.
// ---------------------------------------------------------------------------
__global__ void __launch_bounds__(128)
fc_head(const bf16_t* __restrict__ hseq,   // [BT, H] (layer-2 output)
        const bf16_t* __restrict__ Wfc,    // [1008, H] bf16, rows >=1000 zero
        const float*  __restrict__ bfc, float* __restrict__ out) {
  const int lane = threadIdx.x & 31;
  const int wave = blockIdx.x * 4 + (threadIdx.x >> 5);
  const int mt = wave & 3;
  const int nt = wave >> 2;
  if (nt >= 63) return;
  const int m0 = mt << 4, n0 = nt << 4;
  const int r = lane & 15, kb = (lane & 16) ? 8 : 0;

  v8f acc = {};
#pragma unroll
  for (int k0 = 0; k0 < H_SZ; k0 += 32) {
    // A: h at t = T-1, batch row m0+r, row stride T*H
    const bf16_t* qa = hseq + ((size_t)(m0 + r) * T_SZ + (T_SZ - 1)) * H_SZ + k0 + kb;
    union { v16bf v; v8bf h[2]; } ua;
    ua.h[0] = *(const v8bf*)qa;
    ua.h[1] = *(const v8bf*)(qa + 16);
    const bf16_t* qb = Wfc + (size_t)(n0 + r) * H_SZ + k0 + kb;
    union { v16bf v; v8bf h[2]; } ub;
    ub.h[0] = *(const v8bf*)qb;
    ub.h[1] = *(const v8bf*)(qb + 16);
    acc = wmma_bf16(ua.v, ub.v, acc);
  }
  const int n = n0 + (lane & 15);
  if (n < C_SZ) {
    const float bv = bfc[n];
    const int mr = m0 + ((lane >> 4) << 3);
#pragma unroll
    for (int rr = 0; rr < 8; ++rr)
      out[(size_t)(mr + rr) * C_SZ + n] = acc[rr] + bv;
  }
}

// ---------------------------------------------------------------------------
// Host orchestration (graph-capture safe: only async ops on `stream`)
// ---------------------------------------------------------------------------
extern "C" void kernel_launch(void* const* d_in, const int* in_sizes, int n_in,
                              void* d_out, int out_size, void* d_ws, size_t ws_size,
                              hipStream_t stream) {
  (void)in_sizes; (void)n_in; (void)out_size; (void)ws_size;
  const float* x       = (const float*)d_in[0];
  const float* W_ih[3] = {(const float*)d_in[1], (const float*)d_in[5], (const float*)d_in[9]};
  const float* W_hh[3] = {(const float*)d_in[2], (const float*)d_in[6], (const float*)d_in[10]};
  const float* b_ih[3] = {(const float*)d_in[3], (const float*)d_in[7], (const float*)d_in[11]};
  const float* b_hh[3] = {(const float*)d_in[4], (const float*)d_in[8], (const float*)d_in[12]};
  const float* W_fc    = (const float*)d_in[13];
  const float* b_fc    = (const float*)d_in[14];
  float* out = (float*)d_out;

  // Workspace carve-out (256B aligned slices)
  char* p = (char*)d_ws;
  auto carve = [&](size_t bytes) -> char* {
    char* r = p; p += (bytes + 255) & ~(size_t)255; return r;
  };
  unsigned* bar  = (unsigned*)carve(256);
  float*   xg    = (float*)  carve((size_t)BT * G_SZ * sizeof(float));   // 256 MB
  bf16_t*  hseqA = (bf16_t*) carve((size_t)BT * H_SZ * sizeof(bf16_t));  // 32 MB
  bf16_t*  hseqB = (bf16_t*) carve((size_t)BT * H_SZ * sizeof(bf16_t));  // 32 MB
  bf16_t*  hbuf  = (bf16_t*) carve((size_t)2 * B_SZ * H_SZ * sizeof(bf16_t));
  bf16_t*  x_bf  = (bf16_t*) carve((size_t)BT * I_SZ * sizeof(bf16_t));  // 16 MB
  bf16_t*  wih_bf[3]; bf16_t* whh_bf[3]; float* bias[3];
  for (int l = 0; l < 3; ++l) {
    wih_bf[l] = (bf16_t*)carve((size_t)G_SZ * H_SZ * sizeof(bf16_t));
    whh_bf[l] = (bf16_t*)carve((size_t)G_SZ * H_SZ * sizeof(bf16_t));
    bias[l]   = (float*) carve((size_t)G_SZ * sizeof(float));
  }
  bf16_t* wfc_bf = (bf16_t*)carve((size_t)1008 * H_SZ * sizeof(bf16_t));

  hipMemsetAsync(bar, 0, 256, stream);
  hipMemsetAsync(wfc_bf, 0, (size_t)1008 * H_SZ * sizeof(bf16_t), stream);

  auto cvt = [&](const float* src, bf16_t* dst, int n) {
    cvt_f32_bf16<<<(n + 255) / 256, 256, 0, stream>>>(src, dst, n);
  };

  cvt(x, x_bf, BT * I_SZ);
  const int Kin[3] = {I_SZ, H_SZ, H_SZ};
  for (int l = 0; l < 3; ++l) {
    cvt(W_ih[l], wih_bf[l], G_SZ * Kin[l]);
    cvt(W_hh[l], whh_bf[l], G_SZ * H_SZ);
    bias_sum<<<(G_SZ + 255) / 256, 256, 0, stream>>>(b_ih[l], b_hh[l], bias[l], G_SZ);
  }
  cvt(W_fc, wfc_bf, C_SZ * H_SZ);

  const int gemmBlocks = (BT / 32) * (G_SZ / 64) / 4;   // 8192 blocks of 4 waves
  const bf16_t* Ain = x_bf;
  bf16_t* hs[2] = {hseqA, hseqB};
  int cur = 0;
  for (int l = 0; l < 3; ++l) {
    if (l == 0)
      gemm_xg<I_SZ><<<gemmBlocks, 128, 0, stream>>>(Ain, wih_bf[l], bias[l], xg);
    else
      gemm_xg<H_SZ><<<gemmBlocks, 128, 0, stream>>>(Ain, wih_bf[l], bias[l], xg);
    lstm_scan<<<SCAN_WGS, 128, 0, stream>>>(whh_bf[l], xg, hs[cur], hbuf, bar);
    Ain = hs[cur];
    cur ^= 1;
  }
  fc_head<<<63, 128, 0, stream>>>(Ain, wfc_bf, b_fc, out);
}
